// LlamaAttention_55886114456281
// MI455X (gfx1250) — matrix operand order repaired and checked by reading, not compiled, not run on hardware
//
#include <hip/hip_runtime.h>

#define S_LEN 2048
#define HID_  2048
#define NH    32
#define NKVH  8
#define HD    64
#define GRP   4        // NH / NKVH
#define KVW   (NKVH * HD)   // 512

typedef __bf16 bf16_t;
typedef __attribute__((ext_vector_type(16))) __bf16 bf16x16;
typedef __attribute__((ext_vector_type(8)))  __bf16 bf16x8;
typedef __attribute__((ext_vector_type(8)))  float  f32x8;

__device__ __forceinline__ f32x8 wmma_bf16(bf16x16 a, bf16x16 b, f32x8 c) {
  // D = A(16x32 bf16) x B(32x16 bf16) + C(16x16 f32)
  return __builtin_amdgcn_wmma_f32_16x16x32_bf16(
      /*neg_a=*/false, a, /*neg_b=*/false, b,
      /*c_mod=*/(short)0, c, /*reuse_a=*/false, /*reuse_b=*/false);
}

// A fragment: 16x32 tile of row-major [.. , lda] matrix, rows m0..m0+15, k0..k0+31.
// Lane l (l<16):  row=l,    K = {k0..k0+15}
// Lane l (l>=16): row=l-16, K = {k0+16..k0+31}
__device__ __forceinline__ bf16x16 load_frag_a(const bf16_t* __restrict__ A, int lda,
                                               int m0, int k0, int lane) {
  const int m  = lane & 15;
  const int hi = lane >> 4;
  const bf16_t* p = A + (size_t)(m0 + m) * lda + k0 + 16 * hi;
  return *(const bf16x16*)p;
}

// B fragment for C = X * W^T style ops: stored matrix is [cols, K] row-major
// (rows indexed by output column n, contiguous along contraction K).
// Lane l: col n = n0 + (l&15); elements e=0..15 -> K = k0 + e + 16*(l>=16).
__device__ __forceinline__ bf16x16 load_frag_b_wt(const bf16_t* __restrict__ W, int ldw,
                                                  int n0, int k0, int lane) {
  const int n  = n0 + (lane & 15);
  const int hi = lane >> 4;
  return *(const bf16x16*)(W + (size_t)n * ldw + k0 + 16 * hi);
}

__global__ void cast_f32_bf16(const float* __restrict__ in, bf16_t* __restrict__ out, int n) {
  int i = blockIdx.x * blockDim.x + threadIdx.x;
  if (i < n) out[i] = (bf16_t)in[i];
}

// C[m,n] = sum_k X[m,k] * W[n,k].  X: [M,Kd] bf16, W: [N,Kd] bf16.
// One wave (32 threads) computes a 32x64 tile: 2 A fragments x 4 B fragments
// per k-step -> 12 b128 loads per 8 WMMAs (B fragments amortized over 2 rows).
// Ob != null  : bf16 output, row-major [M,N] (or [N,M] if trans_store)
// Of != null  : fp32 output, row-major [M,N]
__global__ void gemm_bf16_wt(const bf16_t* __restrict__ X,
                             const bf16_t* __restrict__ W,
                             bf16_t* __restrict__ Ob,
                             float* __restrict__ Of,
                             int M, int N, int Kd, int trans_store) {
  const int lane = threadIdx.x;
  const int n0 = blockIdx.x * 64;
  const int m0 = blockIdx.y * 32;

  f32x8 acc[2][4] = {};
  for (int k0 = 0; k0 < Kd; k0 += 32) {
    bf16x16 a0 = load_frag_a(X, Kd, m0,      k0, lane);
    bf16x16 a1 = load_frag_a(X, Kd, m0 + 16, k0, lane);
#pragma unroll
    for (int j = 0; j < 4; ++j) {
      bf16x16 b = load_frag_b_wt(W, Kd, n0 + 16 * j, k0, lane);
      acc[0][j] = wmma_bf16(a0, b, acc[0][j]);
      acc[1][j] = wmma_bf16(a1, b, acc[1][j]);
    }
  }

  const int col = lane & 15;
  const int hi  = lane >> 4;
#pragma unroll
  for (int i = 0; i < 2; ++i) {
#pragma unroll
    for (int j = 0; j < 4; ++j) {
#pragma unroll
      for (int r = 0; r < 8; ++r) {
        const int m = m0 + 16 * i + r + 8 * hi;
        const int n = n0 + 16 * j + col;
        if (Of) {
          Of[(size_t)m * N + n] = acc[i][j][r];
        } else if (trans_store) {
          Ob[(size_t)n * M + m] = (bf16_t)acc[i][j][r];
        } else {
          Ob[(size_t)m * N + n] = (bf16_t)acc[i][j][r];
        }
      }
    }
  }
}

// Flash attention, one wave per (head, 16-query tile).
// Q:  [S, NH*HD] bf16  (head h at cols h*64..)
// Km: [S, KVW]   bf16  (kv-head at cols kvh*64..)
// Vt: [KVW, S]   bf16  (row = kvh*64 + d, contiguous along s)
__global__ void attn_fwd(const bf16_t* __restrict__ Q,
                         const bf16_t* __restrict__ Km,
                         const bf16_t* __restrict__ Vt,
                         const int* __restrict__ amask,
                         bf16_t* __restrict__ out) {
  const int lane = threadIdx.x;
  const int q0  = blockIdx.x * 16;
  const int h   = blockIdx.y;
  const int kvh = h / GRP;

  const bf16_t* Qh = Q  + h   * HD;                   // ld = HID_
  const bf16_t* Kh = Km + kvh * HD;                   // ld = KVW, rows = key pos
  const bf16_t* Vh = Vt + (size_t)(kvh * HD) * S_LEN; // ld = S_LEN, rows = d

  const int col = lane & 15;
  const int hi  = lane >> 4;

  // Q fragments (reused for every key block)
  bf16x16 qa0 = load_frag_a(Qh, HID_, q0, 0,  lane);
  bf16x16 qa1 = load_frag_a(Qh, HID_, q0, 32, lane);

  f32x8 o0 = {}, o1 = {}, o2 = {}, o3 = {};
  float mrow[8], lrow[8];
#pragma unroll
  for (int r = 0; r < 8; ++r) { mrow[r] = -3.0e38f; lrow[r] = 0.0f; }

  __shared__ bf16_t Pl[16 * 32];
  const float scale = 0.125f; // 1/sqrt(64)

  for (int kb = 0; kb < q0 + 16; kb += 32) {
    f32x8 z = {};
    // scores for key cols kb..kb+15 and kb+16..kb+31 (contract over d=0..63)
    f32x8 s0 = wmma_bf16(qa0, load_frag_b_wt(Kh, KVW, kb,      0,  lane), z);
    s0       = wmma_bf16(qa1, load_frag_b_wt(Kh, KVW, kb,      32, lane), s0);
    f32x8 s1 = wmma_bf16(qa0, load_frag_b_wt(Kh, KVW, kb + 16, 0,  lane), z);
    s1       = wmma_bf16(qa1, load_frag_b_wt(Kh, KVW, kb + 16, 32, lane), s1);

    const int key0 = kb + col;
    const int key1 = kb + 16 + col;
    const bool ok0 = (amask[key0] != 0);
    const bool ok1 = (amask[key1] != 0);

#pragma unroll
    for (int r = 0; r < 8; ++r) {
      const int q = q0 + r + 8 * hi;
      float v0 = (ok0 && key0 <= q) ? s0[r] * scale : -3.0e38f;
      float v1 = (ok1 && key1 <= q) ? s1[r] * scale : -3.0e38f;

      // row max over the 32 keys of this block (16 lanes share a row)
      float mx = fmaxf(v0, v1);
      mx = fmaxf(mx, __shfl_xor(mx, 1, 32));
      mx = fmaxf(mx, __shfl_xor(mx, 2, 32));
      mx = fmaxf(mx, __shfl_xor(mx, 4, 32));
      mx = fmaxf(mx, __shfl_xor(mx, 8, 32));

      const float mnew = fmaxf(mrow[r], mx);
      const float corr = __expf(mrow[r] - mnew);
      const float p0 = __expf(v0 - mnew);
      const float p1 = __expf(v1 - mnew);

      float ps = p0 + p1;
      ps += __shfl_xor(ps, 1, 32);
      ps += __shfl_xor(ps, 2, 32);
      ps += __shfl_xor(ps, 4, 32);
      ps += __shfl_xor(ps, 8, 32);

      lrow[r] = lrow[r] * corr + ps;
      mrow[r] = mnew;
      o0[r] *= corr; o1[r] *= corr; o2[r] *= corr; o3[r] *= corr;

      const int m = r + 8 * hi;
      Pl[m * 32 + col]      = (bf16_t)p0;
      Pl[m * 32 + 16 + col] = (bf16_t)p1;
    }
    __syncthreads();

    // P (16x32, contract over keys) x Vt (B: rows=keys via contiguous s)
    bf16x16 pa = load_frag_a(Pl, 32, 0, 0, lane);
    o0 = wmma_bf16(pa, load_frag_b_wt(Vh, S_LEN, 0,  kb, lane), o0);
    o1 = wmma_bf16(pa, load_frag_b_wt(Vh, S_LEN, 16, kb, lane), o1);
    o2 = wmma_bf16(pa, load_frag_b_wt(Vh, S_LEN, 32, kb, lane), o2);
    o3 = wmma_bf16(pa, load_frag_b_wt(Vh, S_LEN, 48, kb, lane), o3);
    __syncthreads();
  }

#pragma unroll
  for (int r = 0; r < 8; ++r) {
    const float inv = 1.0f / lrow[r];
    const int m = q0 + r + 8 * hi;
    bf16_t* dst = out + (size_t)m * HID_ + h * HD;
    dst[col]      = (bf16_t)(o0[r] * inv);
    dst[16 + col] = (bf16_t)(o1[r] * inv);
    dst[32 + col] = (bf16_t)(o2[r] * inv);
    dst[48 + col] = (bf16_t)(o3[r] * inv);
  }
}

extern "C" void kernel_launch(void* const* d_in, const int* in_sizes, int n_in,
                              void* d_out, int out_size, void* d_ws, size_t ws_size,
                              hipStream_t stream) {
  (void)in_sizes; (void)n_in; (void)out_size; (void)ws_size;

  const float* hid   = (const float*)d_in[0];
  const int*   amask = (const int*)  d_in[1];
  const float* Wq    = (const float*)d_in[2];
  const float* Wk    = (const float*)d_in[3];
  const float* Wv    = (const float*)d_in[4];
  const float* Wo    = (const float*)d_in[5];
  float* out = (float*)d_out;

  bf16_t* ws = (bf16_t*)d_ws;
  size_t off = 0;
  bf16_t* Xb  = ws + off; off += (size_t)S_LEN * HID_;   // 4M
  bf16_t* Wqb = ws + off; off += (size_t)HID_ * HID_;    // 4M
  bf16_t* Wkb = ws + off; off += (size_t)KVW * HID_;     // 1M
  bf16_t* Wvb = ws + off; off += (size_t)KVW * HID_;     // 1M
  bf16_t* Wob = ws + off; off += (size_t)HID_ * HID_;    // 4M
  bf16_t* Qb  = ws + off; off += (size_t)S_LEN * HID_;   // 4M
  bf16_t* Kb  = ws + off; off += (size_t)S_LEN * KVW;    // 1M
  bf16_t* Vtb = ws + off; off += (size_t)KVW * S_LEN;    // 1M
  bf16_t* Ab  = ws + off; off += (size_t)S_LEN * HID_;   // 4M  (~48MB total)

  auto cast = [&](const float* src, bf16_t* dst, size_t n) {
    cast_f32_bf16<<<dim3((unsigned)((n + 255) / 256)), dim3(256), 0, stream>>>(src, dst, (int)n);
  };
  cast(hid, Xb,  (size_t)S_LEN * HID_);
  cast(Wq,  Wqb, (size_t)HID_ * HID_);
  cast(Wk,  Wkb, (size_t)KVW * HID_);
  cast(Wv,  Wvb, (size_t)KVW * HID_);
  cast(Wo,  Wob, (size_t)HID_ * HID_);

  const dim3 wave(32, 1, 1);
  // Q = X Wq^T : [S, 2048]
  gemm_bf16_wt<<<dim3(HID_ / 64, S_LEN / 32), wave, 0, stream>>>(
      Xb, Wqb, Qb, nullptr, S_LEN, HID_, HID_, 0);
  // K = X Wk^T : [S, 512]
  gemm_bf16_wt<<<dim3(KVW / 64, S_LEN / 32), wave, 0, stream>>>(
      Xb, Wkb, Kb, nullptr, S_LEN, KVW, HID_, 0);
  // V = X Wv^T, stored transposed: Vt [512, S]
  gemm_bf16_wt<<<dim3(KVW / 64, S_LEN / 32), wave, 0, stream>>>(
      Xb, Wvb, Vtb, nullptr, S_LEN, KVW, HID_, 1);
  // attention -> Ab [S, 2048] bf16
  attn_fwd<<<dim3(S_LEN / 16, NH), wave, 0, stream>>>(Qb, Kb, Vtb, amask, Ab);
  // out = A Wo^T : fp32 [S, 2048]
  gemm_bf16_wt<<<dim3(HID_ / 64, S_LEN / 32), wave, 0, stream>>>(
      Ab, Wob, nullptr, out, S_LEN, HID_, HID_, 0);
}
